// SplitUpsampled_77429670412426
// MI455X (gfx1250) — matrix operand order repaired and checked by reading, not compiled, not run on hardware
//
#include <hip/hip_runtime.h>
#include <hip/hip_bf16.h>
#include <math.h>

typedef __bf16 bf16_t;
typedef bf16_t v16bf __attribute__((ext_vector_type(16)));
typedef bf16_t v8bf  __attribute__((ext_vector_type(8)));
typedef float  v8f   __attribute__((ext_vector_type(8)));
typedef unsigned int u32x4 __attribute__((ext_vector_type(4)));
typedef int          i32x8 __attribute__((ext_vector_type(8)));
typedef int          i32x4 __attribute__((ext_vector_type(4)));

#define NFEAT   96
#define COUT    192
#define BATCH   8
#define HIN     128
#define WIN     128
#define CSH     48          // channels after pixel shuffle
#define HOUT    256
#define WOUT    256
#define KDIM    864         // 96*3*3
#define NPIX    (BATCH*HIN*WIN)          // 131072
#define NIMG    (BATCH*CSH)              // 384
#define IMGSZ   (HOUT*WOUT)              // 65536

#define CONV_COLS 68        // 64 pixels + 2 halo, padded to 68
#define USE_ASYNC_LDS 1     // inline-asm global_load_async_to_lds_b128 fallback

#if defined(__has_builtin)
#if __has_builtin(__builtin_amdgcn_tensor_load_to_lds) && __has_builtin(__builtin_amdgcn_s_wait_tensorcnt)
#define HAVE_TDM 1
#endif
#endif
#ifndef HAVE_TDM
#define HAVE_TDM 0
#endif

// ---------------------------------------------------------------- init d[t]
__global__ void init_d_kernel(float* __restrict__ dre, float* __restrict__ dim) {
    int n = threadIdx.x;                 // 0..255
    float re = 0.f, im = 0.f;
    const float w0 = 6.2831853071795864769f / 256.0f;
    for (int k = -64; k < 64; ++k) {
        float ang = w0 * (float)(k * n);
        re += __cosf(ang);
        im += __sinf(ang);
    }
    dre[n] = re * (1.0f / 256.0f);
    dim[n] = im * (1.0f / 256.0f);
}

// ------------------------------------------------- build circulant L (bf16)
__global__ void init_L_kernel(const float* __restrict__ dre, const float* __restrict__ dim,
                              bf16_t* __restrict__ Lre, bf16_t* __restrict__ Lim,
                              bf16_t* __restrict__ LimN) {
    int idx = blockIdx.x * blockDim.x + threadIdx.x;   // 0..65535
    int m = idx >> 8, n = idx & 255;
    int t = (m - n) & 255;
    Lre[idx]  = (bf16_t)dre[t];
    Lim[idx]  = (bf16_t)dim[t];
    LimN[idx] = (bf16_t)(-dim[t]);
}

// ------------------------------------------------------ f32 -> bf16 convert
__global__ void cvt_kernel(const float* __restrict__ x, const float* __restrict__ w,
                           bf16_t* __restrict__ xb, bf16_t* __restrict__ wb,
                           int nx, int nw) {
    int i = blockIdx.x * blockDim.x + threadIdx.x;
    if (i < nx) xb[i] = (bf16_t)x[i];
    else if (i < nx + nw) wb[i - nx] = (bf16_t)w[i - nx];
}

// ---------------------------------------------- conv 3x3 as implicit GEMM
// Block: 4 waves = 64 consecutive pixels of one row; x slab staged in LDS
// (zero-padded halo). K loop fully unrolled so the k->(ci,r,s) decode
// constant-folds into immediate LDS offsets.
__global__ void __launch_bounds__(128)
conv_wmma_kernel(const bf16_t* __restrict__ xb, const bf16_t* __restrict__ wb,
                 float* __restrict__ yout, bf16_t* __restrict__ yb) {
    __shared__ bf16_t lds_x[NFEAT * 3 * CONV_COLS];    // 38.25 KB

    const int tid   = threadIdx.x;
    const int lane  = tid & 31;
    const int wave  = tid >> 5;
    const int lhalf = lane >> 4;
    const int lmod  = lane & 15;
    const int m0   = blockIdx.x * 16;                  // Cout tile
    const int nblk = blockIdx.y * 64;                  // block pixel base
    const int b  = nblk >> 14;
    const int h  = (nblk & 16383) >> 7;
    const int wb0 = nblk & 127;                        // 0 or 64

    // ---- stage x slab: [ci][r][col], col = -1..66 relative to wb0, zero-padded
    for (int i = tid; i < NFEAT * 3 * CONV_COLS; i += 128) {
        int ci  = i / (3 * CONV_COLS);
        int rem = i - ci * (3 * CONV_COLS);
        int r   = rem / CONV_COLS;
        int col = rem - r * CONV_COLS;
        int hh = h + r - 1;
        int ww = wb0 + col - 1;
        unsigned short val = 0;
        if ((unsigned)hh < (unsigned)HIN && (unsigned)ww < (unsigned)WIN)
            val = *(const unsigned short*)(xb + (((b * NFEAT + ci) * HIN + hh) * WIN + ww));
        ((unsigned short*)lds_x)[i] = val;
    }
    __syncthreads();

    const int wlocal = wave * 16 + lmod;               // lane's pixel within block (B operand N)
    const bf16_t* arow = wb + (m0 + lmod) * KDIM;      // A-matrix row
    v8f acc = {};

    #pragma unroll
    for (int k0 = 0; k0 < KDIM; k0 += 32) {
        // A operand: lanes<16 hold K {k0..+7, k0+16..+23}; lanes>=16 shifted by 8
        const int c0 = k0 + (lhalf ? 8 : 0);
        union { v16bf v; v8bf h2[2]; } A;
        A.h2[0] = *(const v8bf*)(arow + c0);
        A.h2[1] = *(const v8bf*)(arow + c0 + 16);
        // B operand from LDS: element e = K (kb+e+(e&8)), N = wlocal
        const int kb = k0 + (lhalf ? 8 : 0);
        union { v16bf v; unsigned short u[16]; } B;
        #pragma unroll
        for (int e = 0; e < 16; ++e) {
            int k  = kb + e + (e & 8);                 // compile-time except lhalf select
            int ci = k / 9;
            int rs = k - ci * 9;
            int r  = rs / 3;
            int s  = rs - r * 3;
            B.u[e] = ((const unsigned short*)lds_x)[(ci * 3 + r) * CONV_COLS + wlocal + s];
        }
        acc = __builtin_amdgcn_wmma_f32_16x16x32_bf16(false, A.v, false, B.v,
                                                      (short)0, acc, false, false);
    }

    // epilogue: pixel-shuffle scatter; f32 -> d_out (y term), bf16 -> yb
    const int wp = wb0 + wlocal;
    #pragma unroll
    for (int v = 0; v < 8; ++v) {
        int m  = m0 + v + 8 * lhalf;
        int c  = m >> 2;
        int i2 = (m >> 1) & 1;
        int j2 = m & 1;
        int idx = ((b * CSH + c) * HOUT + (2 * h + i2)) * WOUT + (2 * wp + j2);
        float f = acc[v];
        yout[idx] = f;
        yb[idx]   = (bf16_t)f;
    }
}

// ------------------------------------- row transform: T1 = (Lre + i Lim) * Y
// Block: 4 waves share a 64-pixel column span; per k-step the 32x64 Y tile is
// DMA'd to LDS by the Tensor Data Mover (wave 0 issues, all waves barrier),
// falling back to async-to-LDS copies if the TDM builtin is unavailable.
__global__ void __launch_bounds__(128)
dft_rows_kernel(const bf16_t* __restrict__ yb,
                const bf16_t* __restrict__ Lre, const bf16_t* __restrict__ Lim,
                bf16_t* __restrict__ t1re, bf16_t* __restrict__ t1im) {
    __shared__ bf16_t lds_y[32 * 64];                  // 4 KB tile

    const int tid   = threadIdx.x;
    const int lane  = tid & 31;
    const int wave  = tid >> 5;
    const int lhalf = lane >> 4;
    const int lmod  = lane & 15;
    const int m0   = blockIdx.x * 16;
    const int nblk = blockIdx.y * 64;
    const int img  = blockIdx.z;
    const bf16_t* Y  = yb + (size_t)img * IMGSZ;
    const bf16_t* ar = Lre + (m0 + lmod) * 256;
    const bf16_t* ai = Lim + (m0 + lmod) * 256;
    const int nloc = wave * 16 + lmod;                 // lane's column within the 64-wide tile
    const unsigned lbase = (unsigned)(size_t)(&lds_y[0]);

#if !HAVE_TDM
    // fallback staging assignment: thread copies 32 bytes of one row
    const int srow = tid >> 2;                         // 0..31
    const int scol = (tid & 3) * 16;                   // 0,16,32,48
    const unsigned ldst0 = lbase + (unsigned)(srow * 64 + scol) * 2u;
    const unsigned ldst1 = ldst0 + 16u;
#endif

    v8f cre = {}, cim = {};
    for (int k0 = 0; k0 < 256; k0 += 32) {
        __syncthreads();                               // previous tile fully consumed
#if HAVE_TDM
        if (wave == 0) {
            // D# per CDNA5 ISA: 2-D tile 64x32 elems, 2B each
            unsigned long long ga = (unsigned long long)(size_t)(Y + k0 * 256 + nblk);
            u32x4 g0;
            g0.x = 1u;                                             // count=1 (valid)
            g0.y = lbase;                                          // lds_addr
            g0.z = (unsigned)ga;                                   // global_addr[31:0]
            g0.w = (unsigned)((ga >> 32) & 0x1FFFFFFu) | 0x80000000u; // addr[56:32] | type=2
            i32x8 g1 = {};
            g1[0] = 1 << 16;                                       // data_size = 1 (2 bytes)
            g1[1] = (int)(256u << 16);                             // tensor_dim0 = 256
            g1[2] = 0;                                             // dim0 hi / dim1 lo
            g1[3] = (64 << 16) | 1;                                // tile_dim0=64 | tensor_dim1=65536
            g1[4] = 32;                                            // tile_dim1=32, tile_dim2=0
            g1[5] = 256;                                           // tensor_dim0_stride = 256
            g1[6] = 0;
            g1[7] = 0;
            i32x4 gz4 = {};
            i32x8 gz8 = {};
            __builtin_amdgcn_tensor_load_to_lds(g0, g1, gz4, gz4, gz8, 0);
            __builtin_amdgcn_s_wait_tensorcnt((short)0);
        }
#else
        const bf16_t* g = Y + (k0 + srow) * 256 + nblk + scol;
#if USE_ASYNC_LDS
        asm volatile("global_load_async_to_lds_b128 %0, %1, off"
                     :: "v"(ldst0), "v"(g) : "memory");
        asm volatile("global_load_async_to_lds_b128 %0, %1, off"
                     :: "v"(ldst1), "v"(g + 8) : "memory");
        asm volatile("s_wait_asynccnt 0x0" ::: "memory");
#else
        *(uint4*)((char*)lds_y + (srow * 64 + scol) * 2)      = *(const uint4*)g;
        *(uint4*)((char*)lds_y + (srow * 64 + scol) * 2 + 16) = *(const uint4*)(g + 8);
#endif
#endif
        if (k0 + 32 < 256)
            __builtin_prefetch(Y + (k0 + 32 + (tid >> 2)) * 256 + nblk + (tid & 3) * 16);
        __syncthreads();                               // tile visible to all waves

        const int c0 = k0 + (lhalf ? 8 : 0);
        union { v16bf v; v8bf h2[2]; } Are, Aim;
        Are.h2[0] = *(const v8bf*)(ar + c0);  Are.h2[1] = *(const v8bf*)(ar + c0 + 16);
        Aim.h2[0] = *(const v8bf*)(ai + c0);  Aim.h2[1] = *(const v8bf*)(ai + c0 + 16);

        const int kb = (lhalf ? 8 : 0);
        union { v16bf v; unsigned short u[16]; } B;
        #pragma unroll
        for (int e = 0; e < 16; ++e)
            B.u[e] = ((const unsigned short*)lds_y)[(kb + e + (e & 8)) * 64 + nloc];

        cre = __builtin_amdgcn_wmma_f32_16x16x32_bf16(false, Are.v, false, B.v, (short)0, cre, false, false);
        cim = __builtin_amdgcn_wmma_f32_16x16x32_bf16(false, Aim.v, false, B.v, (short)0, cim, false, false);
    }

    const int ncol = nblk + nloc;
    #pragma unroll
    for (int v = 0; v < 8; ++v) {
        int m = m0 + v + 8 * lhalf;
        size_t idx = (size_t)img * IMGSZ + m * 256 + ncol;
        t1re[idx] = (bf16_t)cre[v];
        t1im[idx] = (bf16_t)cim[v];
    }
}

// ------------- col transform + epilogue: low = |T1 * L^T|, out = (1-2b)low + b*y
__global__ void __launch_bounds__(128)
dft_cols_kernel(const bf16_t* __restrict__ t1re, const bf16_t* __restrict__ t1im,
                const bf16_t* __restrict__ Lre, const bf16_t* __restrict__ Lim,
                const bf16_t* __restrict__ LimN,
                const float* __restrict__ beta_p, float* __restrict__ out) {
    const int lane  = threadIdx.x & 31;
    const int wave  = threadIdx.x >> 5;
    const int lhalf = lane >> 4;
    const int lmod  = lane & 15;
    const int m0  = blockIdx.x * 16;
    const int n0  = (blockIdx.y * 4 + wave) * 16;
    const int img = blockIdx.z;
    const bf16_t* Ar = t1re + (size_t)img * IMGSZ + (m0 + lmod) * 256;
    const bf16_t* Ai = t1im + (size_t)img * IMGSZ + (m0 + lmod) * 256;
    // column n of L^T == row n of L (contiguous in K -> pure vector loads)
    const bf16_t* Br = Lre  + (n0 + lmod) * 256;
    const bf16_t* Bi = Lim  + (n0 + lmod) * 256;
    const bf16_t* Bn = LimN + (n0 + lmod) * 256;

    v8f cre = {}, cim = {};
    for (int k0 = 0; k0 < 256; k0 += 32) {
        const int c0 = k0 + (lhalf ? 16 : 0);
        const int kb = k0 + (lhalf ? 16 : 0);
        union { v16bf v; v8bf h2[2]; } Are, Aim, Brv, Biv, Bnv;
        Are.h2[0] = *(const v8bf*)(Ar + c0);  Are.h2[1] = *(const v8bf*)(Ar + c0 + 8);
        Aim.h2[0] = *(const v8bf*)(Ai + c0);  Aim.h2[1] = *(const v8bf*)(Ai + c0 + 8);
        Brv.h2[0] = *(const v8bf*)(Br + kb);  Brv.h2[1] = *(const v8bf*)(Br + kb + 8);
        Biv.h2[0] = *(const v8bf*)(Bi + kb);  Biv.h2[1] = *(const v8bf*)(Bi + kb + 8);
        Bnv.h2[0] = *(const v8bf*)(Bn + kb);  Bnv.h2[1] = *(const v8bf*)(Bn + kb + 8);
        // low_re = T1re*LreT - T1im*LimT ; low_im = T1re*LimT + T1im*LreT
        cre = __builtin_amdgcn_wmma_f32_16x16x32_bf16(false, Are.v, false, Brv.v, (short)0, cre, false, false);
        cre = __builtin_amdgcn_wmma_f32_16x16x32_bf16(false, Aim.v, false, Bnv.v, (short)0, cre, false, false);
        cim = __builtin_amdgcn_wmma_f32_16x16x32_bf16(false, Are.v, false, Biv.v, (short)0, cim, false, false);
        cim = __builtin_amdgcn_wmma_f32_16x16x32_bf16(false, Aim.v, false, Brv.v, (short)0, cim, false, false);
    }

    const float beta = *beta_p;
    const float a0   = 1.0f - 2.0f * beta;
    #pragma unroll
    for (int v = 0; v < 8; ++v) {
        int m = m0 + v + 8 * lhalf;
        size_t idx = (size_t)img * IMGSZ + m * 256 + (n0 + lmod);
        float lo = sqrtf(cre[v] * cre[v] + cim[v] * cim[v]);
        out[idx] = a0 * lo + beta * out[idx];   // out currently holds y (written by conv)
    }
}

// ----------------------------------------------------------------- launcher
extern "C" void kernel_launch(void* const* d_in, const int* in_sizes, int n_in,
                              void* d_out, int out_size, void* d_ws, size_t ws_size,
                              hipStream_t stream) {
    const float* x    = (const float*)d_in[0];
    const float* w    = (const float*)d_in[1];
    const float* beta = (const float*)d_in[2];
    float* out = (float*)d_out;

    const int NX = BATCH * NFEAT * HIN * WIN;   // 12,582,912
    const int NW = COUT * NFEAT * 9;            // 165,888
    const size_t NY = (size_t)NIMG * IMGSZ;     // 25,165,824

    char* ws = (char*)d_ws;
    size_t o = 0;
    bf16_t* xb   = (bf16_t*)(ws + o); o += (size_t)NX * 2;
    bf16_t* wbuf = (bf16_t*)(ws + o); o += (size_t)NW * 2;
    bf16_t* yb   = (bf16_t*)(ws + o); o += NY * 2;
    bf16_t* t1re = (bf16_t*)(ws + o); o += NY * 2;
    bf16_t* t1im = (bf16_t*)(ws + o); o += NY * 2;
    bf16_t* Lre  = (bf16_t*)(ws + o); o += 65536 * 2;
    bf16_t* Lim  = (bf16_t*)(ws + o); o += 65536 * 2;
    bf16_t* LimN = (bf16_t*)(ws + o); o += 65536 * 2;
    float*  dre  = (float*)(ws + o);  o += 256 * 4;
    float*  dim  = (float*)(ws + o);  o += 256 * 4;

    init_d_kernel<<<1, 256, 0, stream>>>(dre, dim);
    init_L_kernel<<<256, 256, 0, stream>>>(dre, dim, Lre, Lim, LimN);

    int ncvt = NX + NW;
    cvt_kernel<<<(ncvt + 255) / 256, 256, 0, stream>>>(x, w, xb, wbuf, NX, NW);

    // conv: 12 Cout tiles x 2048 blocks (64 pixels each)
    conv_wmma_kernel<<<dim3(COUT / 16, NPIX / 64), 128, 0, stream>>>(xb, wbuf, out, yb);

    // DFT row pass: 16 m-tiles x 4 blocks (64 cols) x 384 images
    dft_rows_kernel<<<dim3(16, 4, NIMG), 128, 0, stream>>>(yb, Lre, Lim, t1re, t1im);

    // DFT col pass + final combine
    dft_cols_kernel<<<dim3(16, 4, NIMG), 128, 0, stream>>>(t1re, t1im, Lre, Lim, LimN, beta, out);
}
